// CrossAttentionBlockWithTime_16398185136215
// MI455X (gfx1250) — matrix operand order repaired and checked by reading, not compiled, not run on hardware
//
#include <hip/hip_runtime.h>
#include <hip/hip_bf16.h>

typedef __attribute__((ext_vector_type(16))) _Float16 v16h;
typedef __attribute__((ext_vector_type(8)))  float    v8f;
typedef __attribute__((ext_vector_type(4)))  int      v4i;

union F16Frag { v16h v; v4i q[2]; _Float16 h[16]; };
union I4H8    { v4i q; _Float16 h[8]; };

__device__ __forceinline__ v8f wmma16(const v16h& a, const v16h& b, const v8f& c) {
  return __builtin_amdgcn_wmma_f32_16x16x32_f16(false, a, false, b, (short)0, c, false, false);
}
__device__ __forceinline__ float sigmoidf_(float x) { return 1.f / (1.f + __expf(-x)); }
__device__ __forceinline__ float siluf_(float x)    { return x * sigmoidf_(x); }

// ---------------------------------------------------------------------------
// f16 WMMA GEMM: C[M,N] = A[M,K] @ B[K,N] + bias (+residual)
// block = 256 threads (8 waves). Tile M=128, N=128, K-step=32, double-buffered
// LDS (one barrier per K-step). Wave w: mw=(w>>1)*32, nw=(w&1)*64 ->
// 2x4 accumulators of 16x16 = 8 WMMAs per K-step per wave.
// ---------------------------------------------------------------------------
__global__ __launch_bounds__(256)
void gemm_wmma_f16(const _Float16* __restrict__ A, const _Float16* __restrict__ Bm,
                   const float* __restrict__ bias, const float* __restrict__ residual,
                   float* __restrict__ outF, _Float16* __restrict__ outH,
                   int M, int N, int K)
{
  __shared__ _Float16 sA[2][128 * 40];   // [row][k], stride 40 halves
  __shared__ _Float16 sBt[2][128 * 40];  // [n][k] (transposed), stride 40 halves
  const int tid  = threadIdx.x;
  const int w    = tid >> 5, lane = tid & 31;
  const int lm   = lane & 15, lh = lane >> 4;
  const int bm0  = blockIdx.y * 128, bn0 = blockIdx.x * 128;
  const int mw   = (w >> 1) * 32, nw = (w & 1) * 64;

  // Per-thread fill coordinates (A: 512 int4, B: 512 int4; 2 each per thread).
  const int liA0 = tid, liA1 = tid + 256;
  const int rA0 = liA0 >> 2, cA0 = liA0 & 3;
  const int rA1 = liA1 >> 2, cA1 = liA1 & 3;
  const int liB0 = tid, liB1 = tid + 256;
  const int rB0 = liB0 >> 4, cB0 = liB0 & 15;
  const int rB1 = liB1 >> 4, cB1 = liB1 & 15;

  const _Float16* pA0 = A + (size_t)(bm0 + rA0) * K + cA0 * 8;
  const _Float16* pA1 = A + (size_t)(bm0 + rA1) * K + cA1 * 8;
  const _Float16* pB0 = Bm + (size_t)rB0 * N + bn0 + cB0 * 8;
  const _Float16* pB1 = Bm + (size_t)rB1 * N + bn0 + cB1 * 8;
  const size_t stepB = (size_t)32 * N;

  // Prologue: stage k0=0 tile into buffer 0.
  {
    v4i ra0 = *(const v4i*)pA0, ra1 = *(const v4i*)pA1;
    I4H8 ub0, ub1;
    ub0.q = *(const v4i*)pB0;
    ub1.q = *(const v4i*)pB1;
    pA0 += 32; pA1 += 32; pB0 += stepB; pB1 += stepB;
    *(v4i*)(&sA[0][rA0 * 40 + cA0 * 8]) = ra0;
    *(v4i*)(&sA[0][rA1 * 40 + cA1 * 8]) = ra1;
#pragma unroll
    for (int e = 0; e < 8; ++e) sBt[0][(cB0 * 8 + e) * 40 + rB0] = ub0.h[e];
#pragma unroll
    for (int e = 0; e < 8; ++e) sBt[0][(cB1 * 8 + e) * 40 + rB1] = ub1.h[e];
  }

  v8f acc[2][4] = {};
  int cur = 0;

  for (int k0 = 0; k0 < K; k0 += 32) {
    __syncthreads();
    const bool hn = (k0 + 32) < K;
    v4i ra0, ra1;
    I4H8 ub0, ub1;
    if (hn) {  // stage next tile (global loads overlap with WMMAs below)
      ra0 = *(const v4i*)pA0;
      ra1 = *(const v4i*)pA1;
      ub0.q = *(const v4i*)pB0;
      ub1.q = *(const v4i*)pB1;
      __builtin_prefetch(pA0 + 32, 0, 3);
      __builtin_prefetch(pB0 + stepB, 0, 3);
    }

    F16Frag a[2], b[4];
#pragma unroll
    for (int mi = 0; mi < 2; ++mi) {
      const _Float16* p = &sA[cur][(mw + mi * 16 + lm) * 40 + lh * 8];
      a[mi].q[0] = *(const v4i*)(p);
      a[mi].q[1] = *(const v4i*)(p + 16);
    }
#pragma unroll
    for (int nj = 0; nj < 4; ++nj) {
      const _Float16* p = &sBt[cur][(nw + nj * 16 + lm) * 40 + lh * 8];
      b[nj].q[0] = *(const v4i*)(p);
      b[nj].q[1] = *(const v4i*)(p + 16);
    }
#pragma unroll
    for (int mi = 0; mi < 2; ++mi)
#pragma unroll
      for (int nj = 0; nj < 4; ++nj)
        acc[mi][nj] = wmma16(a[mi].v, b[nj].v, acc[mi][nj]);

    if (hn) {
      const int nxt = cur ^ 1;
      *(v4i*)(&sA[nxt][rA0 * 40 + cA0 * 8]) = ra0;
      *(v4i*)(&sA[nxt][rA1 * 40 + cA1 * 8]) = ra1;
#pragma unroll
      for (int e = 0; e < 8; ++e) sBt[nxt][(cB0 * 8 + e) * 40 + rB0] = ub0.h[e];
#pragma unroll
      for (int e = 0; e < 8; ++e) sBt[nxt][(cB1 * 8 + e) * 40 + rB1] = ub1.h[e];
      pA0 += 32; pA1 += 32; pB0 += stepB; pB1 += stepB;
    }
    cur ^= 1;
  }

#pragma unroll
  for (int mi = 0; mi < 2; ++mi)
#pragma unroll
    for (int nj = 0; nj < 4; ++nj)
#pragma unroll
      for (int r = 0; r < 8; ++r) {
        int row = bm0 + mw + mi * 16 + lh * 8 + r;
        int col = bn0 + nw + nj * 16 + lm;
        size_t idx = (size_t)row * N + col;
        float v = acc[mi][nj][r] + bias[col];
        if (residual) v += residual[idx];
        if (outF) outF[idx] = v;
        if (outH) outH[idx] = (_Float16)v;
      }
}

// ---------------------------------------------------------------------------
// Flash attention (attends m >= n). q:[B*L,256] f16 (col=h*32+d),
// kv:[B*L,2304] f16 (k at col h*32+d, v at col 256+h*256+d), o:[B*L,2048] f16.
// grid (L/128, H, B), block 256. Each wave owns 16 query rows.
// ---------------------------------------------------------------------------
__global__ __launch_bounds__(256)
void attn_wmma(const _Float16* __restrict__ q, const _Float16* __restrict__ kv,
               _Float16* __restrict__ o)
{
  __shared__ _Float16 sK[32 * 40];    // [key][chan]
  __shared__ _Float16 sVt[256 * 40];  // [dim][key] transposed
  __shared__ _Float16 sP[8 * 16 * 40];// per-wave 16x32 P tiles
  const int tid = threadIdx.x, w = tid >> 5, lane = tid & 31;
  const int lm = lane & 15, lh = lane >> 4;
  const int b = blockIdx.z, hh = blockIdx.y, nb = blockIdx.x;
  const int n0w = nb * 128 + w * 16;
  const int L = 2048, LDKV = 2304;
  const size_t rowbase = (size_t)b * L;

  F16Frag aq;
  {
    const _Float16* p = q + (rowbase + n0w + lm) * 256 + hh * 32 + lh * 8;
    aq.q[0] = *(const v4i*)(p);
    aq.q[1] = *(const v4i*)(p + 16);
  }
  v8f accO[16] = {};
  float mrow[8], lrow[8];
#pragma unroll
  for (int r = 0; r < 8; ++r) { mrow[r] = -1e30f; lrow[r] = 0.f; }
  const float scale = 0.1767766952966369f; // 32^-0.5

  for (int m0 = nb * 128; m0 < L; m0 += 32) {
    if (tid < 128) {  // K block 32x32
      int r = tid >> 2, c = tid & 3;
      v4i d = *(const v4i*)(kv + (rowbase + m0 + r) * LDKV + hh * 32 + c * 8);
      *(v4i*)(&sK[r * 40 + c * 8]) = d;
    }
#pragma unroll
    for (int i = 0; i < 4; ++i) {  // V block 32x256, store transposed
      int li = tid + i * 256;
      int r = li >> 5, c = li & 31;
      I4H8 u; u.q = *(const v4i*)(kv + (rowbase + m0 + r) * LDKV + 256 + hh * 256 + c * 8);
#pragma unroll
      for (int e = 0; e < 8; ++e) sVt[(c * 8 + e) * 40 + r] = u.h[e];
    }
    __syncthreads();

    if (m0 + 32 > n0w) {  // wave-uniform: skip fully masked key blocks
      v8f s[2];
#pragma unroll
      for (int tt = 0; tt < 2; ++tt) {
        F16Frag bk;
        const _Float16* p = &sK[(tt * 16 + lm) * 40 + lh * 8];
        bk.q[0] = *(const v4i*)(p);
        bk.q[1] = *(const v4i*)(p + 16);
        v8f z = {};
        s[tt] = wmma16(aq.v, bk.v, z);
      }
      float rmax[8];
#pragma unroll
      for (int r = 0; r < 8; ++r) {
        int na = n0w + lh * 8 + r;
        float v0 = s[0][r] * scale; if (m0 + lm < na)      v0 = -1e30f;
        float v1 = s[1][r] * scale; if (m0 + 16 + lm < na) v1 = -1e30f;
        s[0][r] = v0; s[1][r] = v1;
        rmax[r] = fmaxf(v0, v1);
      }
#pragma unroll
      for (int msk = 8; msk >= 1; msk >>= 1)
#pragma unroll
        for (int r = 0; r < 8; ++r)
          rmax[r] = fmaxf(rmax[r], __shfl_xor(rmax[r], msk, 32));
      float alpha[8], rsum[8];
#pragma unroll
      for (int r = 0; r < 8; ++r) {
        float mn = fmaxf(mrow[r], rmax[r]);
        alpha[r] = __expf(mrow[r] - mn);
        mrow[r] = mn;
        float p0 = __expf(s[0][r] - mn);
        float p1 = __expf(s[1][r] - mn);
        rsum[r] = p0 + p1;
        _Float16* pp = &sP[w * 640 + (lh * 8 + r) * 40];
        pp[lm]      = (_Float16)p0;
        pp[16 + lm] = (_Float16)p1;
      }
#pragma unroll
      for (int msk = 8; msk >= 1; msk >>= 1)
#pragma unroll
        for (int r = 0; r < 8; ++r)
          rsum[r] += __shfl_xor(rsum[r], msk, 32);
#pragma unroll
      for (int r = 0; r < 8; ++r) lrow[r] = lrow[r] * alpha[r] + rsum[r];
#pragma unroll
      for (int d = 0; d < 16; ++d)
#pragma unroll
        for (int r = 0; r < 8; ++r) accO[d][r] *= alpha[r];

      F16Frag ap;  // re-load P in A-fragment layout (in-wave DS ordering)
      {
        const _Float16* p = &sP[w * 640 + lm * 40 + lh * 8];
        ap.q[0] = *(const v4i*)(p);
        ap.q[1] = *(const v4i*)(p + 16);
      }
#pragma unroll
      for (int d = 0; d < 16; ++d) {
        F16Frag bv;
        const _Float16* p = &sVt[(d * 16 + lm) * 40 + lh * 8];
        bv.q[0] = *(const v4i*)(p);
        bv.q[1] = *(const v4i*)(p + 16);
        accO[d] = wmma16(ap.v, bv.v, accO[d]);
      }
    }
    __syncthreads();
  }

#pragma unroll
  for (int r = 0; r < 8; ++r) {
    float inv = 1.f / lrow[r];
    size_t row = rowbase + n0w + lh * 8 + r;
#pragma unroll
    for (int d = 0; d < 16; ++d)
      o[row * 2048 + hh * 256 + d * 16 + lm] = (_Float16)(accO[d][r] * inv);
  }
}

// ---------------------------------------------------------------------------
// Elementwise / small kernels
// ---------------------------------------------------------------------------
__global__ void cvt_f16_kernel(const float* __restrict__ src, _Float16* __restrict__ dst, int n) {
  for (int i = blockIdx.x * blockDim.x + threadIdx.x; i < n; i += gridDim.x * blockDim.x)
    dst[i] = (_Float16)src[i];
}

// a1 = silu(LN(x)*g+b + pos) ; one wave per 256-wide row
__global__ __launch_bounds__(256)
void prep_q_kernel(const float* __restrict__ x, const float* __restrict__ pos,
                   const float* __restrict__ g, const float* __restrict__ be,
                   _Float16* __restrict__ out)
{
  int w = threadIdx.x >> 5, lane = threadIdx.x & 31;
  int row = blockIdx.x * 8 + w;
  const float* xr = x + (size_t)row * 256;
  float v[8], s = 0.f, s2 = 0.f;
#pragma unroll
  for (int i = 0; i < 8; ++i) { v[i] = xr[lane + i * 32]; s += v[i]; s2 += v[i] * v[i]; }
#pragma unroll
  for (int m = 16; m >= 1; m >>= 1) { s += __shfl_xor(s, m, 32); s2 += __shfl_xor(s2, m, 32); }
  float mean = s * (1.f / 256.f);
  float var  = s2 * (1.f / 256.f) - mean * mean;
  float rstd = rsqrtf(var + 1e-5f);
  int l = row & 2047;
#pragma unroll
  for (int i = 0; i < 8; ++i) {
    int c = lane + i * 32;
    float f = (v[i] - mean) * rstd * g[c] + be[c] + pos[l * 256 + c];
    out[(size_t)row * 256 + c] = (_Float16)siluf_(f);
  }
}

// out = silu(LN(in)*g+b) with f16 in/out, rows of 256
__global__ __launch_bounds__(256)
void ln_silu_f16_kernel(const _Float16* __restrict__ in, const float* __restrict__ g,
                        const float* __restrict__ be, _Float16* __restrict__ out)
{
  int w = threadIdx.x >> 5, lane = threadIdx.x & 31;
  int row = blockIdx.x * 8 + w;
  const _Float16* xr = in + (size_t)row * 256;
  float v[8], s = 0.f, s2 = 0.f;
#pragma unroll
  for (int i = 0; i < 8; ++i) { v[i] = (float)xr[lane + i * 32]; s += v[i]; s2 += v[i] * v[i]; }
#pragma unroll
  for (int m = 16; m >= 1; m >>= 1) { s += __shfl_xor(s, m, 32); s2 += __shfl_xor(s2, m, 32); }
  float mean = s * (1.f / 256.f);
  float var  = s2 * (1.f / 256.f) - mean * mean;
  float rstd = rsqrtf(var + 1e-5f);
#pragma unroll
  for (int i = 0; i < 8; ++i) {
    int c = lane + i * 32;
    float f = (v[i] - mean) * rstd * g[c] + be[c];
    out[(size_t)row * 256 + c] = (_Float16)siluf_(f);
  }
}

// ffa = silu(LN(x2 + t_shift)*g+b), t_shift = th[b,0:256]
__global__ __launch_bounds__(256)
void prep_ff_kernel(const float* __restrict__ x2, const float* __restrict__ th,
                    const float* __restrict__ g, const float* __restrict__ be,
                    _Float16* __restrict__ out)
{
  int w = threadIdx.x >> 5, lane = threadIdx.x & 31;
  int row = blockIdx.x * 8 + w;
  int bb = row >> 11;
  const float* xr = x2 + (size_t)row * 256;
  float v[8], s = 0.f, s2 = 0.f;
#pragma unroll
  for (int i = 0; i < 8; ++i) {
    int c = lane + i * 32;
    v[i] = xr[c] + th[bb * 1280 + c];
    s += v[i]; s2 += v[i] * v[i];
  }
#pragma unroll
  for (int m = 16; m >= 1; m >>= 1) { s += __shfl_xor(s, m, 32); s2 += __shfl_xor(s2, m, 32); }
  float mean = s * (1.f / 256.f);
  float var  = s2 * (1.f / 256.f) - mean * mean;
  float rstd = rsqrtf(var + 1e-5f);
#pragma unroll
  for (int i = 0; i < 8; ++i) {
    int c = lane + i * 32;
    float f = (v[i] - mean) * rstd * g[c] + be[c];
    out[(size_t)row * 256 + c] = (_Float16)siluf_(f);
  }
}

// g = silu(h * sigmoid(t_scale)), t_scale = th[b,256:1280], h:[8192,1024]
__global__ void gate_kernel(const _Float16* __restrict__ h, const float* __restrict__ th,
                            _Float16* __restrict__ out)
{
  const int n = 8192 * 1024;
  for (int i = blockIdx.x * blockDim.x + threadIdx.x; i < n; i += gridDim.x * blockDim.x) {
    int row = i >> 10, col = i & 1023, bb = row >> 11;
    float z = (float)h[i] * sigmoidf_(th[bb * 1280 + 256 + col]);
    out[i] = (_Float16)siluf_(z);
  }
}

__global__ void time_mlp1_kernel(const float* __restrict__ t, const float* __restrict__ Wt1,
                                 const float* __restrict__ bt1, float* __restrict__ th1)
{
  int b = blockIdx.x, j = threadIdx.x;
  float acc = bt1[j];
  for (int i = 0; i < 256; ++i) acc += t[b * 256 + i] * Wt1[i * 256 + j];
  th1[b * 256 + j] = siluf_(acc);
}

__global__ void time_mlp2_kernel(const float* __restrict__ th1, const float* __restrict__ Wt2,
                                 const float* __restrict__ bt2, float* __restrict__ th)
{
  int b = blockIdx.y;
  int col = blockIdx.x * 256 + threadIdx.x;
  float acc = bt2[col];
  for (int i = 0; i < 256; ++i) acc += th1[b * 256 + i] * Wt2[i * 1280 + col];
  th[b * 1280 + col] = acc;
}

// ---------------------------------------------------------------------------
extern "C" void kernel_launch(void* const* d_in, const int* in_sizes, int n_in,
                              void* d_out, int out_size, void* d_ws, size_t ws_size,
                              hipStream_t stream) {
  const float* x      = (const float*)d_in[0];
  const float* ctx    = (const float*)d_in[1];
  const float* t      = (const float*)d_in[2];
  const float* pos    = (const float*)d_in[3];
  const float* ln_g   = (const float*)d_in[4];
  const float* ln_b   = (const float*)d_in[5];
  const float* Wq     = (const float*)d_in[6];
  const float* bq     = (const float*)d_in[7];
  const float* Wkv1   = (const float*)d_in[8];
  const float* bkv1   = (const float*)d_in[9];
  const float* kvln_g = (const float*)d_in[10];
  const float* kvln_b = (const float*)d_in[11];
  const float* Wkv2   = (const float*)d_in[12];
  const float* bkv2   = (const float*)d_in[13];
  const float* Wm     = (const float*)d_in[14];
  const float* bm     = (const float*)d_in[15];
  const float* Wt1    = (const float*)d_in[16];
  const float* bt1    = (const float*)d_in[17];
  const float* Wt2    = (const float*)d_in[18];
  const float* bt2    = (const float*)d_in[19];
  const float* ff1g   = (const float*)d_in[20];
  const float* ff1b   = (const float*)d_in[21];
  const float* Wff1   = (const float*)d_in[22];
  const float* bff1   = (const float*)d_in[23];
  const float* Wff2   = (const float*)d_in[24];
  const float* bff2   = (const float*)d_in[25];
  float* outp = (float*)d_out;

  char* wp = (char*)d_ws;
  auto take = [&](size_t bytes) -> void* {
    void* r = (void*)wp;
    wp += (bytes + 255) & ~(size_t)255;
    return r;
  };
  const size_t R = 8192;  // B*L rows
  _Float16* Wqh   = (_Float16*)take((size_t)65536 * 2);
  _Float16* Wkv1h = (_Float16*)take((size_t)65536 * 2);
  _Float16* Wkv2h = (_Float16*)take((size_t)589824 * 2);
  _Float16* Wmh   = (_Float16*)take((size_t)524288 * 2);
  _Float16* Wff1h = (_Float16*)take((size_t)262144 * 2);
  _Float16* Wff2h = (_Float16*)take((size_t)262144 * 2);
  _Float16* ctxh  = (_Float16*)take(R * 256 * 2);
  _Float16* a1h   = (_Float16*)take(R * 256 * 2);
  _Float16* qh    = (_Float16*)take(R * 256 * 2);
  _Float16* kvt   = (_Float16*)take(R * 256 * 2);
  _Float16* kva   = (_Float16*)take(R * 256 * 2);
  _Float16* kvh   = (_Float16*)take(R * 2304 * 2);
  _Float16* aoh   = (_Float16*)take(R * 2048 * 2);
  _Float16* hhb   = (_Float16*)take(R * 1024 * 2);
  _Float16* ghb   = (_Float16*)take(R * 1024 * 2);
  float*    x2    = (float*)take(R * 256 * 4);
  float*    th1   = (float*)take(1024 * 4);
  float*    th    = (float*)take(5120 * 4);

  // Weight / context conversion to f16
  cvt_f16_kernel<<<256, 256, 0, stream>>>(Wq,   Wqh,   65536);
  cvt_f16_kernel<<<256, 256, 0, stream>>>(Wkv1, Wkv1h, 65536);
  cvt_f16_kernel<<<1024, 256, 0, stream>>>(Wkv2, Wkv2h, 589824);
  cvt_f16_kernel<<<1024, 256, 0, stream>>>(Wm,   Wmh,   524288);
  cvt_f16_kernel<<<512, 256, 0, stream>>>(Wff1, Wff1h, 262144);
  cvt_f16_kernel<<<512, 256, 0, stream>>>(Wff2, Wff2h, 262144);
  cvt_f16_kernel<<<2048, 256, 0, stream>>>(ctx, ctxh, 2097152);

  // q = silu(LN(x)+pos) @ Wq + bq
  prep_q_kernel<<<1024, 256, 0, stream>>>(x, pos, ln_g, ln_b, a1h);
  gemm_wmma_f16<<<dim3(256 / 128, 8192 / 128), 256, 0, stream>>>(
      a1h, Wqh, bq, nullptr, nullptr, qh, 8192, 256, 256);

  // kv = silu(LN(context @ Wkv1 + bkv1)) @ Wkv2 + bkv2
  gemm_wmma_f16<<<dim3(256 / 128, 8192 / 128), 256, 0, stream>>>(
      ctxh, Wkv1h, bkv1, nullptr, nullptr, kvt, 8192, 256, 256);
  ln_silu_f16_kernel<<<1024, 256, 0, stream>>>(kvt, kvln_g, kvln_b, kva);
  gemm_wmma_f16<<<dim3(2304 / 128, 8192 / 128), 256, 0, stream>>>(
      kva, Wkv2h, bkv2, nullptr, nullptr, kvh, 8192, 2304, 256);

  // flash attention (attends m >= n)
  attn_wmma<<<dim3(16, 8, 4), 256, 0, stream>>>(qh, kvh, aoh);

  // x2 = x + attn_out @ Wm + bm
  gemm_wmma_f16<<<dim3(256 / 128, 8192 / 128), 256, 0, stream>>>(
      aoh, Wmh, bm, x, x2, nullptr, 8192, 256, 2048);

  // time MLP: th = silu(t@Wt1+bt1) @ Wt2 + bt2
  time_mlp1_kernel<<<4, 256, 0, stream>>>(t, Wt1, bt1, th1);
  time_mlp2_kernel<<<dim3(5, 4), 256, 0, stream>>>(th1, Wt2, bt2, th);

  // h = silu(LN(x2 + t_shift)) @ Wff1 + bff1
  prep_ff_kernel<<<1024, 256, 0, stream>>>(x2, th, ff1g, ff1b, a1h);
  gemm_wmma_f16<<<dim3(1024 / 128, 8192 / 128), 256, 0, stream>>>(
      a1h, Wff1h, bff1, nullptr, nullptr, hhb, 8192, 1024, 256);

  // out = silu(h * sigmoid(t_scale)) @ Wff2 + bff2 + x2
  gate_kernel<<<4096, 256, 0, stream>>>(hhb, th, ghb);
  gemm_wmma_f16<<<dim3(256 / 128, 8192 / 128), 256, 0, stream>>>(
      ghb, Wff2h, bff2, x2, outp, nullptr, 8192, 256, 1024);
}
